// PoseRPN_90855738180396
// MI455X (gfx1250) — compile-verified, matrix operand order checked by reading, not run on hardware
//
#include <hip/hip_runtime.h>
#include <hip/hip_bf16.h>
#include <math.h>

typedef __attribute__((ext_vector_type(16))) _Float16 v16h;
typedef __attribute__((ext_vector_type(8)))  float    v8f;

#define IMGW 1344.0f
#define IMGH 800.0f
#define NTOT 3350
#define NPAD 4096
#define POSTN 100

// ---------------------------------------------------------------------------
// Repack fp32 conv weights [256,256,3,3] into f16 WMMA B-fragments:
// wpk[((t*8+kc)*16+nt)*512 + lane*16 + e]
//   n  = nt*16 + (lane&15)
//   k  = e + 8*(e>>3) + 8*(lane>>4)      (16-bit B 32x16 per-lane layout)
//   ci = kc*32 + k ; tap t = ky*3+kx
// ---------------------------------------------------------------------------
__global__ __launch_bounds__(256) void pack_weights(const float* __restrict__ w,
                                                    _Float16* __restrict__ wpk) {
    int idx = blockIdx.x * 256 + threadIdx.x;           // 9*8*16*512 = 589824
    if (idx >= 9 * 8 * 16 * 512) return;
    int e    = idx & 15;
    int lane = (idx >> 4) & 31;
    int nt   = (idx >> 9) & 15;
    int kc   = (idx >> 13) & 7;
    int t    = idx >> 16;
    int n  = nt * 16 + (lane & 15);
    int k  = e + 8 * (e >> 3) + 8 * (lane >> 4);
    int ci = kc * 32 + k;
    wpk[idx] = (_Float16)w[((size_t)n * 256 + ci) * 9 + t];
}

// ---------------------------------------------------------------------------
// 3x3 conv, C=256 -> 256, implicit GEMM via v_wmma_f32_16x16x32_f16.
// Block = 256 thr (8 waves). One 16-position M-tile; each wave -> 2 cout tiles.
// A tile double-buffered in LDS, pre-swizzled into per-lane fragment layout.
// Staging threads mapped position-major for coalesced global loads; all
// spatial math hoisted; K-loop unrolled by 8 so tap math is per-tap not
// per-step.
// ---------------------------------------------------------------------------
__global__ __launch_bounds__(256) void conv3x3_wmma(const float* __restrict__ in,
                                                    const _Float16* __restrict__ wpk,
                                                    const float* __restrict__ bias,
                                                    float* __restrict__ out,
                                                    int H, int W, int HW) {
    const int img   = blockIdx.z;
    const int ptile = blockIdx.x;
    const int tid   = threadIdx.x;
    const int wave  = tid >> 5;
    const int lane  = tid & 31;

    in  += (size_t)img * 256 * HW;
    out += (size_t)img * 256 * HW;

    __shared__ __align__(32) _Float16 ldsA[2][512];    // double-buffered frag

    v8f acc0 = {}, acc1 = {};
    const int nt0 = wave * 2, nt1 = wave * 2 + 1;
    const int p0  = ptile * 16;

    // ---- per-thread staging geometry (loop invariant) ----
    // Thread -> (position r, channels kA and kA+16). Consecutive tids cover
    // consecutive positions of one channel -> coalesced 64B bursts.
    const int r  = tid & 15;
    const int kA = tid >> 4;                           // 0..15
    const int p  = p0 + r;
    const int y  = p / W, x = p - y * W;
    const bool inP = p < HW;
    // fragment swizzle: lane = r + 16*(kA>>3), e = kA&7 (elem B: e+8)
    const int offA = (r + 16 * (kA >> 3)) * 16 + (kA & 7);
    const int offB = offA + 8;
    const size_t chA = (size_t)kA * HW;
    const size_t ch16 = (size_t)16 * HW;

    auto fillStep = [&](int it, int buf) {
        const int t  = it >> 3;                        // tap 0..8
        const int kc = it & 7;                         // 32-ch chunk 0..7
        const int t3 = (t * 11) >> 5;                  // t/3 for t<9
        const int dy = t3 - 1, dx = t - t3 * 3 - 1;
        const int yy = y + dy, xx = x + dx;
        float vA = 0.f, vB = 0.f;
        if (inP && (unsigned)yy < (unsigned)H && (unsigned)xx < (unsigned)W) {
            const size_t base = (size_t)(kc * 32) * HW + chA + yy * W + xx;
            vA = in[base];
            vB = in[base + ch16];
        }
        ldsA[buf][offA] = (_Float16)vA;
        ldsA[buf][offB] = (_Float16)vB;
    };

    fillStep(0, 0);
    __syncthreads();

    const _Float16* wb0 = wpk + (size_t)nt0 * 512 + lane * 16;
    const _Float16* wb1 = wpk + (size_t)nt1 * 512 + lane * 16;

#pragma unroll 8
    for (int it = 0; it < 72; ++it) {
        const int buf = it & 1;
        if (it + 1 < 72) fillStep(it + 1, buf ^ 1);    // overlap fill w/ compute
        v16h a  = *(const v16h*)&ldsA[buf][lane * 16];
        v16h b0 = *(const v16h*)(wb0 + (size_t)it * 8192);
        v16h b1 = *(const v16h*)(wb1 + (size_t)it * 8192);
        __builtin_prefetch(wb0 + (size_t)(it + 1) * 8192, 0, 1);
        acc0 = __builtin_amdgcn_wmma_f32_16x16x32_f16(false, a, false, b0,
                                                      (short)0, acc0, false, false);
        acc1 = __builtin_amdgcn_wmma_f32_16x16x32_f16(false, a, false, b1,
                                                      (short)0, acc1, false, false);
        __syncthreads();
    }

    // D layout: lane<16 -> M=r, lane>=16 -> M=r+8; N = lane&15
    int co0 = nt0 * 16 + (lane & 15);
    int co1 = nt1 * 16 + (lane & 15);
    int rb  = p0 + 8 * (lane >> 4);
    float b0v = bias[co0], b1v = bias[co1];
    for (int rr = 0; rr < 8; ++rr) {
        int pp = rb + rr;
        if (pp < HW) {
            out[(size_t)co0 * HW + pp] = acc0[rr] + b0v;
            out[(size_t)co1 * HW + pp] = acc1[rr] + b1v;
        }
    }
}

// ---------------------------------------------------------------------------
// GroupNorm(32 groups) + ReLU. One block per (group, image).
// ---------------------------------------------------------------------------
__global__ __launch_bounds__(256) void groupnorm_relu(const float* __restrict__ in,
                                                      float* __restrict__ out,
                                                      const float* __restrict__ gamma,
                                                      const float* __restrict__ beta,
                                                      int HW) {
    int g = blockIdx.x, img = blockIdx.z;
    const float* src = in  + ((size_t)img * 256 + g * 8) * HW;
    float*       dst = out + ((size_t)img * 256 + g * 8) * HW;
    int n = 8 * HW;
    float s = 0.f, ss = 0.f;
    for (int i = threadIdx.x; i < n; i += 256) { float v = src[i]; s += v; ss += v * v; }
    __shared__ float r0[256], r1[256];
    r0[threadIdx.x] = s; r1[threadIdx.x] = ss;
    __syncthreads();
    for (int o = 128; o > 0; o >>= 1) {
        if (threadIdx.x < o) { r0[threadIdx.x] += r0[threadIdx.x + o];
                               r1[threadIdx.x] += r1[threadIdx.x + o]; }
        __syncthreads();
    }
    float mean = r0[0] / n;
    float var  = r1[0] / n - mean * mean;
    float inv  = rsqrtf(var + 1e-5f);
    for (int i = threadIdx.x; i < n; i += 256) {
        int c = g * 8 + i / HW;
        float v = (src[i] - mean) * inv * gamma[c] + beta[c];
        dst[i] = v > 0.f ? v : 0.f;
    }
}

// ---------------------------------------------------------------------------
// Small-output 3x3 head conv (Cout in {1,4}).
// ---------------------------------------------------------------------------
__global__ __launch_bounds__(256) void head_conv(const float* __restrict__ in,
                                                 const float* __restrict__ w,
                                                 const float* __restrict__ bias,
                                                 float* __restrict__ out,
                                                 int Cout, int H, int W, int HW) {
    int p = blockIdx.x * 256 + threadIdx.x;
    int co = blockIdx.y, img = blockIdx.z;
    if (p >= HW) return;
    int y = p / W, x = p - y * W;
    const float* src = in + (size_t)img * 256 * HW;
    float acc = bias[co];
    for (int t = 0; t < 9; ++t) {
        int yy = y + t / 3 - 1, xx = x + t % 3 - 1;
        if ((unsigned)yy < (unsigned)H && (unsigned)xx < (unsigned)W) {
            const float* wp = w + (size_t)co * 2304 + t;
            const float* ip = src + yy * W + xx;
            for (int ci = 0; ci < 256; ++ci)
                acc += ip[(size_t)ci * HW] * wp[(size_t)ci * 9];
        }
    }
    out[((size_t)img * Cout + co) * HW + p] = acc;
}

// sqrt(sigmoid(cls)*sigmoid(ctr)), masked by cls>0.05
__global__ __launch_bounds__(256) void score_kernel(const float* __restrict__ logits,
                                                    const float* __restrict__ ctr,
                                                    float* __restrict__ score, int HW) {
    int p = blockIdx.x * 256 + threadIdx.x;
    int img = blockIdx.z;
    if (p >= HW) return;
    float cs = 1.f / (1.f + expf(-logits[(size_t)img * HW + p]));
    float ts = 1.f / (1.f + expf(-ctr[(size_t)img * HW + p]));
    float sc = sqrtf(cs * ts);
    score[(size_t)img * HW + p] = (cs > 0.05f) ? sc : 0.f;
}

// ---------------------------------------------------------------------------
// Per-image top-k (iterative block max) + box decode into global slots.
// ---------------------------------------------------------------------------
__global__ __launch_bounds__(1024) void topk_decode(const float* __restrict__ score,
                                                    const float* __restrict__ reg,
                                                    const float* __restrict__ scales,
                                                    float* __restrict__ oboxes,
                                                    float* __restrict__ oscores,
                                                    float* __restrict__ stmp,
                                                    int HW, int W, int stride,
                                                    int k, int off, int lvl) {
    int img = blockIdx.x;
    const float* s = score + (size_t)img * HW;
    float* sw = stmp + (size_t)img * HW;
    for (int i = threadIdx.x; i < HW; i += 1024) sw[i] = s[i];
    __syncthreads();
    __shared__ float bval[1024];
    __shared__ int   bidx[1024];
    float scale = scales[lvl];
    for (int sel = 0; sel < k; ++sel) {
        float best = -1e30f; int bi = 0;
        for (int i = threadIdx.x; i < HW; i += 1024) {
            float v = sw[i];
            if (v > best) { best = v; bi = i; }
        }
        bval[threadIdx.x] = best; bidx[threadIdx.x] = bi;
        __syncthreads();
        for (int o = 512; o > 0; o >>= 1) {
            if (threadIdx.x < o && bval[threadIdx.x + o] > bval[threadIdx.x]) {
                bval[threadIdx.x] = bval[threadIdx.x + o];
                bidx[threadIdx.x] = bidx[threadIdx.x + o];
            }
            __syncthreads();
        }
        if (threadIdx.x == 0) {
            int i = bidx[0];
            float sc = bval[0];
            sw[i] = -2e30f;
            int y = i / W, x = i - y * W;
            float lx = (float)(x * stride + stride / 2);
            float ly = (float)(y * stride + stride / 2);
            const float* r = reg + (size_t)img * 4 * HW;
            float l  = expf(scale * r[i]);
            float tt = expf(scale * r[HW + i]);
            float rr = expf(scale * r[2 * (size_t)HW + i]);
            float bb = expf(scale * r[3 * (size_t)HW + i]);
            float x1 = fminf(fmaxf(lx - l,  0.f), IMGW - 1.f);
            float y1 = fminf(fmaxf(ly - tt, 0.f), IMGH - 1.f);
            float x2 = fminf(fmaxf(lx + rr, 0.f), IMGW - 1.f);
            float y2 = fminf(fmaxf(ly + bb, 0.f), IMGH - 1.f);
            int slot = off + sel;
            float* ob = oboxes + ((size_t)img * NTOT + slot) * 4;
            ob[0] = x1; ob[1] = y1; ob[2] = x2; ob[3] = y2;
            oscores[(size_t)img * NTOT + slot] = sc;
        }
        __syncthreads();
    }
}

// ---------------------------------------------------------------------------
// Per-image NMS: bitonic sort (desc) of 4096 padded entries in LDS, greedy
// suppression (+1 IoU convention), write [100,5] output.
// ---------------------------------------------------------------------------
__global__ __launch_bounds__(1024) void nms_kernel(const float* __restrict__ boxes,
                                                   const float* __restrict__ scores,
                                                   float* __restrict__ outp) {
    int img = blockIdx.x;
    __shared__ float ss[NPAD];
    __shared__ int   si[NPAD];
    __shared__ unsigned char keep[NPAD];
    for (int i = threadIdx.x; i < NPAD; i += 1024) {
        ss[i] = (i < NTOT) ? scores[(size_t)img * NTOT + i] : -1e30f;
        si[i] = i;
        keep[i] = (i < NTOT) ? 1 : 0;
    }
    __syncthreads();
    for (int ksz = 2; ksz <= NPAD; ksz <<= 1) {
        for (int j = ksz >> 1; j > 0; j >>= 1) {
            for (int i = threadIdx.x; i < NPAD; i += 1024) {
                int ixj = i ^ j;
                if (ixj > i) {
                    bool up = ((i & ksz) == 0);   // descending blocks
                    bool sw = up ? (ss[i] < ss[ixj]) : (ss[i] > ss[ixj]);
                    if (sw) {
                        float tv = ss[i]; ss[i] = ss[ixj]; ss[ixj] = tv;
                        int ti = si[i]; si[i] = si[ixj]; si[ixj] = ti;
                    }
                }
            }
            __syncthreads();
        }
    }
    const float* bb = boxes + (size_t)img * NTOT * 4;
    for (int i = 0; i < NTOT; ++i) {
        if (keep[i]) {
            int ii = si[i];
            float ax1 = bb[ii * 4], ay1 = bb[ii * 4 + 1];
            float ax2 = bb[ii * 4 + 2], ay2 = bb[ii * 4 + 3];
            float aarea = (ax2 - ax1 + 1.f) * (ay2 - ay1 + 1.f);
            for (int j = i + 1 + threadIdx.x; j < NTOT; j += 1024) {
                if (keep[j]) {
                    int jj = si[j];
                    float bx1 = bb[jj * 4], by1 = bb[jj * 4 + 1];
                    float bx2 = bb[jj * 4 + 2], by2 = bb[jj * 4 + 3];
                    float barea = (bx2 - bx1 + 1.f) * (by2 - by1 + 1.f);
                    float iw = fminf(ax2, bx2) - fmaxf(ax1, bx1) + 1.f;
                    float ih = fminf(ay2, by2) - fmaxf(ay1, by1) + 1.f;
                    iw = fmaxf(iw, 0.f); ih = fmaxf(ih, 0.f);
                    float inter = iw * ih;
                    float iou = inter / (aarea + barea - inter);
                    if (iou > 0.6f) keep[j] = 0;
                }
            }
        }
        __syncthreads();
    }
    if (threadIdx.x == 0) {
        int cnt = 0;
        for (int i = 0; i < NTOT && cnt < POSTN; ++i) {
            if (keep[i] && ss[i] > 0.f) {
                int ii = si[i];
                float* o = outp + ((size_t)img * POSTN + cnt) * 5;
                o[0] = bb[ii * 4]; o[1] = bb[ii * 4 + 1];
                o[2] = bb[ii * 4 + 2]; o[3] = bb[ii * 4 + 3];
                o[4] = ss[i];
                ++cnt;
            }
        }
        for (; cnt < POSTN; ++cnt) {
            float* o = outp + ((size_t)img * POSTN + cnt) * 5;
            o[0] = o[1] = o[2] = o[3] = o[4] = 0.f;
        }
    }
}

// ---------------------------------------------------------------------------
// Host orchestration
// ---------------------------------------------------------------------------
extern "C" void kernel_launch(void* const* d_in, const int* in_sizes, int n_in,
                              void* d_out, int out_size, void* d_ws, size_t ws_size,
                              hipStream_t stream) {
    (void)in_sizes; (void)n_in; (void)out_size; (void)ws_size;
    static const int HS[5] = {100, 50, 25, 13, 7};
    static const int WS[5] = {168, 84, 42, 21, 11};
    static const int STR[5] = {8, 16, 32, 64, 128};
    const int B = 2;

    // jax pytree flatten order (sorted dict keys):
    // feats[5], bbox_b, bbox_tower(4x{b,beta,gamma,w}), bbox_w, cls_b,
    // cls_tower(4x{b,beta,gamma,w}), cls_w, ctr_b, ctr_w, scales
    const float* feats[5];
    for (int i = 0; i < 5; ++i) feats[i] = (const float*)d_in[i];
    const float* bbox_b = (const float*)d_in[5];
    struct Lyr { const float *b, *beta, *gamma, *w; };
    Lyr bt[4], ct[4];
    for (int l = 0; l < 4; ++l) {
        bt[l].b = (const float*)d_in[6 + 4 * l];  bt[l].beta  = (const float*)d_in[7 + 4 * l];
        bt[l].gamma = (const float*)d_in[8 + 4 * l]; bt[l].w = (const float*)d_in[9 + 4 * l];
    }
    const float* bbox_w = (const float*)d_in[22];
    const float* cls_b  = (const float*)d_in[23];
    for (int l = 0; l < 4; ++l) {
        ct[l].b = (const float*)d_in[24 + 4 * l]; ct[l].beta  = (const float*)d_in[25 + 4 * l];
        ct[l].gamma = (const float*)d_in[26 + 4 * l]; ct[l].w = (const float*)d_in[27 + 4 * l];
    }
    const float* cls_w  = (const float*)d_in[40];
    const float* ctr_b  = (const float*)d_in[41];
    const float* ctr_w  = (const float*)d_in[42];
    const float* scales = (const float*)d_in[43];

    // workspace layout
    const size_t WPK_ELEMS = (size_t)9 * 8 * 16 * 512;   // halfs per layer (589824)
    const int HWMAX = 100 * 168;
    char* ws = (char*)d_ws;
    _Float16* wpk = (_Float16*)ws;                ws += 8 * WPK_ELEMS * sizeof(_Float16);
    float* bufA   = (float*)ws;                   ws += (size_t)B * 256 * HWMAX * 4;
    float* bufB   = (float*)ws;                   ws += (size_t)B * 256 * HWMAX * 4;
    float* logits = (float*)ws;                   ws += (size_t)B * HWMAX * 4;
    float* ctrb   = (float*)ws;                   ws += (size_t)B * HWMAX * 4;
    float* regb   = (float*)ws;                   ws += (size_t)B * 4 * HWMAX * 4;
    float* scoreb = (float*)ws;                   ws += (size_t)B * HWMAX * 4;
    float* stmp   = (float*)ws;                   ws += (size_t)B * HWMAX * 4;
    float* oboxes = (float*)ws;                   ws += (size_t)B * NTOT * 4 * 4;
    float* oscor  = (float*)ws;                   ws += (size_t)B * NTOT * 4;

    // pack weights (cls tower -> slots 0..3, bbox tower -> slots 4..7)
    const int packBlocks = (int)(WPK_ELEMS / 256);
    for (int l = 0; l < 4; ++l) {
        pack_weights<<<packBlocks, 256, 0, stream>>>(ct[l].w, wpk + (size_t)l * WPK_ELEMS);
        pack_weights<<<packBlocks, 256, 0, stream>>>(bt[l].w, wpk + (size_t)(4 + l) * WPK_ELEMS);
    }

    int off = 0;
    for (int lvl = 0; lvl < 5; ++lvl) {
        const int H = HS[lvl], W = WS[lvl], HW = H * W;
        const int mt = (HW + 15) / 16;
        dim3 cg(mt, 1, B);
        dim3 gg(32, 1, B);
        dim3 hg((HW + 255) / 256, 1, B);

        // cls tower
        const float* src = feats[lvl];
        for (int l = 0; l < 4; ++l) {
            conv3x3_wmma<<<cg, 256, 0, stream>>>(src, wpk + (size_t)l * WPK_ELEMS,
                                                 ct[l].b, bufA, H, W, HW);
            groupnorm_relu<<<gg, 256, 0, stream>>>(bufA, bufB, ct[l].gamma, ct[l].beta, HW);
            src = bufB;
        }
        head_conv<<<dim3((HW + 255) / 256, 1, B), 256, 0, stream>>>(bufB, cls_w, cls_b,
                                                                    logits, 1, H, W, HW);
        // bbox tower
        src = feats[lvl];
        for (int l = 0; l < 4; ++l) {
            conv3x3_wmma<<<cg, 256, 0, stream>>>(src, wpk + (size_t)(4 + l) * WPK_ELEMS,
                                                 bt[l].b, bufA, H, W, HW);
            groupnorm_relu<<<gg, 256, 0, stream>>>(bufA, bufB, bt[l].gamma, bt[l].beta, HW);
            src = bufB;
        }
        head_conv<<<dim3((HW + 255) / 256, 4, B), 256, 0, stream>>>(bufB, bbox_w, bbox_b,
                                                                    regb, 4, H, W, HW);
        head_conv<<<dim3((HW + 255) / 256, 1, B), 256, 0, stream>>>(bufB, ctr_w, ctr_b,
                                                                    ctrb, 1, H, W, HW);
        score_kernel<<<hg, 256, 0, stream>>>(logits, ctrb, scoreb, HW);

        int k = HW < 1000 ? HW : 1000;
        topk_decode<<<B, 1024, 0, stream>>>(scoreb, regb, scales, oboxes, oscor, stmp,
                                            HW, W, STR[lvl], k, off, lvl);
        off += k;
    }

    nms_kernel<<<B, 1024, 0, stream>>>(oboxes, oscor, (float*)d_out);
}
